// TEM_67508295959408
// MI455X (gfx1250) — matrix operand order repaired and checked by reading, not compiled, not run on hardware
//
#include <hip/hip_runtime.h>
#include <cstdint>
#include <cstddef>

// ---------------- problem constants ----------------
#define S_     128
#define B_     256
#define G_     128
#define PHASE_ 256
#define SS_    512
#define SH_    1024
#define NA_    4
#define DM_    32
#define G2_    256          // 2*G
#define DINP_  288          // 258 padded up to multiple of 32
#define DH_    1280         // 10*G
#define BETA_  5.0f
#define BETA0_ 10.986122886681098f   // log(0.9/0.1)*5
#define EPS_   1e-8f

typedef __attribute__((ext_vector_type(16))) __bf16 v16bf;
typedef __attribute__((ext_vector_type(8)))  float  v8f;

union BFFrag { v16bf v; unsigned u[8]; };

__device__ inline unsigned short f2bf(float f) {
  unsigned u = __float_as_uint(f);
  u += 0x7FFFu + ((u >> 16) & 1u);       // round-to-nearest-even
  return (unsigned short)(u >> 16);
}

// ---------------- block reductions (blockDim.x == 128) ----------------
__device__ inline float blk_sum(float v, volatile float* red) {
  int tid = threadIdx.x;
  red[tid] = v; __syncthreads();
  for (int s = 64; s > 0; s >>= 1) {
    if (tid < s) red[tid] = red[tid] + red[tid + s];
    __syncthreads();
  }
  float r = red[0]; __syncthreads(); return r;
}
__device__ inline float blk_max(float v, volatile float* red) {
  int tid = threadIdx.x;
  red[tid] = v; __syncthreads();
  for (int s = 64; s > 0; s >>= 1) {
    if (tid < s) red[tid] = fmaxf(red[tid], red[tid + s]);
    __syncthreads();
  }
  float r = red[0]; __syncthreads(); return r;
}

// ---------------- bf16 WMMA GEMM: C(256xN) = act(A(256xK,f32) * W(KxN) + bias) ----------------
// W pre-converted to bf16, N-major: WT[n*K + k]. K % 32 == 0, N % 64 == 0.
// Block 256 threads = 8 waves (wave32); block tile 128x64; wave tile 32x32 (2x2 WMMA).
// LDS pitch 18 dwords: conflict-free fragment reads + 8B-aligned packed stores.
// A tile: float4 global loads -> packed bf16 b64 LDS stores.
// W tile: GLOBAL_LOAD_ASYNC_TO_LDS_B64 (CDNA5 async DMA, ASYNCcnt-tracked).
#define LDP 18
template<int ACT>   // 0 = none, 1 = ELU, 2 = sigmoid
__global__ __launch_bounds__(256) void gemm_bf16(
    const float* __restrict__ A, const unsigned short* __restrict__ WT,
    const float* __restrict__ bias, float* __restrict__ C, int N, int K)
{
  __shared__ __align__(16) unsigned As32[128 * LDP];  // 128 rows x 32 bf16
  __shared__ __align__(16) unsigned Bs32[64 * LDP];   // 64 cols(n) x 32 bf16 (k-major in row)
  unsigned short* Bs16 = (unsigned short*)Bs32;

  const int tid  = threadIdx.x;
  const int lane = tid & 31;
  const int w    = tid >> 5;
  const int wm   = w >> 1;              // 0..3 -> m offset wm*32
  const int wn   = w & 1;               // 0..1 -> n offset wn*32
  const int lm   = lane & 15;
  const int kh   = lane >> 4;           // lane half
  const int bm   = blockIdx.y * 128;
  const int bn   = blockIdx.x * 64;

  v8f acc[2][2];
  #pragma unroll
  for (int i = 0; i < 2; ++i)
    #pragma unroll
    for (int j = 0; j < 2; ++j) {
      v8f z = {0.f,0.f,0.f,0.f,0.f,0.f,0.f,0.f};
      acc[i][j] = z;
    }

  for (int kb = 0; kb < K; kb += 32) {
    // ---- W tile via async DMA: 64 rows x 32 bf16 = 512 x b64, 2 per thread ----
    #pragma unroll
    for (int it = 0; it < 2; ++it) {
      int e = tid + it * 256;
      int n = e >> 3, part = e & 7;                       // 8 x b64 per 64B row
      unsigned lds_off = (unsigned)(uintptr_t)(Bs16 + n * (2 * LDP) + part * 4);
      unsigned long long gsrc =
          (unsigned long long)(uintptr_t)(WT + (size_t)(bn + n) * K + kb + part * 4);
      asm volatile("global_load_async_to_lds_b64 %0, %1, off"
                   :: "v"(lds_off), "v"(gsrc) : "memory");
    }
    // ---- A tile: 128 rows x 32 f32 -> bf16, b128 loads / b64 LDS stores ----
    #pragma unroll
    for (int it = 0; it < 4; ++it) {
      int e = tid + it * 256;                              // 1024 float4 chunks
      int r = e >> 3, c4 = (e & 7) << 2;                   // c4 = 0,4,...,28
      const float4 va = *(const float4*)(A + (size_t)(bm + r) * K + kb + c4);
      unsigned lo = ((unsigned)f2bf(va.y) << 16) | (unsigned)f2bf(va.x);
      unsigned hi = ((unsigned)f2bf(va.w) << 16) | (unsigned)f2bf(va.z);
      unsigned long long pack = ((unsigned long long)hi << 32) | (unsigned long long)lo;
      *(unsigned long long*)(As32 + r * LDP + (c4 >> 1)) = pack;
    }
    asm volatile("s_wait_asynccnt 0x0" ::: "memory");
    __syncthreads();

    BFFrag af[2], bf[2];
    #pragma unroll
    for (int mt = 0; mt < 2; ++mt) {            // A frags: row = M, interleaved K pairs
      int row = wm * 32 + mt * 16 + lm;
      #pragma unroll
      for (int v = 0; v < 8; ++v) {
        int dw = (v & 3) + ((v >> 2) << 3) + (kh << 2);
        af[mt].u[v] = As32[row * LDP + dw];
      }
    }
    #pragma unroll
    for (int nt = 0; nt < 2; ++nt) {            // B frags: row = N, interleaved K pairs
      int row = wn * 32 + nt * 16 + lm;
      #pragma unroll
      for (int v = 0; v < 8; ++v) {
        int dw = (v & 3) + ((v >> 2) << 3) + (kh << 2);
        bf[nt].u[v] = Bs32[row * LDP + dw];
      }
    }
    #pragma unroll
    for (int mt = 0; mt < 2; ++mt)
      #pragma unroll
      for (int nt = 0; nt < 2; ++nt)
        acc[mt][nt] = __builtin_amdgcn_wmma_f32_16x16x32_bf16(
            false, af[mt].v, false, bf[nt].v, (short)0, acc[mt][nt], false, false);
    __syncthreads();
  }

  #pragma unroll
  for (int mt = 0; mt < 2; ++mt) {
    #pragma unroll
    for (int nt = 0; nt < 2; ++nt) {
      int col = bn + wn * 32 + nt * 16 + lm;
      float bv = bias ? bias[col] : 0.f;
      #pragma unroll
      for (int v = 0; v < 8; ++v) {
        int row = bm + wm * 32 + mt * 16 + kh * 8 + v;   // C/D: VGPR v -> M = v + 8*lanehalf
        float xv = acc[mt][nt][v] + bv;
        if (ACT == 1)      xv = xv > 0.f ? xv : (__expf(xv) - 1.f);
        else if (ACT == 2) xv = 1.f / (1.f + __expf(-xv));
        C[(size_t)row * N + col] = xv;
      }
    }
  }
}

// ---------------- masked softmax retrieval (one block per batch element) ----------------
template<int DQ, int DV, bool WITH_ENT, bool WITH_ERR>
__global__ __launch_bounds__(128) void retrieve_k(
    const float* __restrict__ Q, const float* __restrict__ KEYS,
    const float* __restrict__ VALS, float* __restrict__ RET,
    float* __restrict__ ENT, float* __restrict__ ERR,
    const float* __restrict__ XREF, int icur)
{
  __shared__ __align__(16) float qs[DQ];
  __shared__ float ps[S_ + 1];
  __shared__ float red[128];
  __shared__ float s_mx, s_sum;
  const int b = blockIdx.x, tid = threadIdx.x;

  for (int c = tid; c < DQ; c += 128) qs[c] = Q[(size_t)b * DQ + c];
  __syncthreads();

  float lg = -1e9f;
  if (tid < icur) {
    const float4* kr4 = (const float4*)(KEYS + ((size_t)b * S_ + tid) * DQ);
    float acc = 0.f;
    #pragma unroll 2
    for (int c = 0; c < DQ / 4; ++c) {
      float4 kv = kr4[c];
      acc = fmaf(qs[4 * c + 0], kv.x, acc);
      acc = fmaf(qs[4 * c + 1], kv.y, acc);
      acc = fmaf(qs[4 * c + 2], kv.z, acc);
      acc = fmaf(qs[4 * c + 3], kv.w, acc);
    }
    lg = BETA_ * acc;
  }
  float mx = blk_max(lg, red);
  if (tid == 0) s_mx = fmaxf(mx, BETA0_);
  __syncthreads();
  float m = s_mx;
  float e = __expf(lg - m);
  float ssum = blk_sum(e, red);
  if (tid == 0) s_sum = ssum + __expf(BETA0_ - m);
  __syncthreads();
  float inv = 1.f / s_sum;
  ps[tid] = e * inv;
  if (tid == 0) ps[S_] = __expf(BETA0_ - m) * inv;
  __syncthreads();

  if (WITH_ENT) {
    float p = ps[tid];
    float h = p * __logf(p + EPS_);
    if (tid == 0) { float pn = ps[S_]; h += pn * __logf(pn + EPS_); }
    float tot = blk_sum(h, red);
    if (tid == 0) ENT[b] = -tot;
  }

  float errloc = 0.f;
  for (int d4 = tid; d4 < DV / 4; d4 += 128) {
    float4 a = {0.f, 0.f, 0.f, 0.f};
    for (int s = 0; s < icur; ++s) {
      float p = ps[s];
      float4 vv = *(const float4*)(VALS + ((size_t)b * S_ + s) * DV + 4 * d4);
      a.x = fmaf(p, vv.x, a.x); a.y = fmaf(p, vv.y, a.y);
      a.z = fmaf(p, vv.z, a.z); a.w = fmaf(p, vv.w, a.w);
    }
    *(float4*)(RET + (size_t)b * DV + 4 * d4) = a;
    if (WITH_ERR) {
      float4 xr = *(const float4*)(XREF + (size_t)b * DV + 4 * d4);
      float dx = xr.x - a.x, dy = xr.y - a.y, dz = xr.z - a.z, dw = xr.w - a.w;
      errloc += dx * dx + dy * dy + dz * dz + dw * dw;
    }
  }
  if (WITH_ERR) {
    float tot = blk_sum(errloc, red);
    if (tid == 0) ERR[b] = tot;
  }
}

// ---------------- small fused kernels ----------------
// h = tanh(d_t @ w1); Abuf[b, k*G+j] = h[k] * g[b,j]
__global__ __launch_bounds__(128) void phaseA_k(
    const float* __restrict__ d_t, const float* __restrict__ w1,
    const float* __restrict__ g, float* __restrict__ Abuf)
{
  __shared__ float h[DM_];
  int b = blockIdx.x, tid = threadIdx.x;
  if (tid < DM_) {
    float a = 0.f;
    #pragma unroll
    for (int r = 0; r < NA_; ++r) a = fmaf(d_t[(size_t)b * NA_ + r], w1[r * DM_ + tid], a);
    h[tid] = tanhf(a);
  }
  __syncthreads();
  float gj = g[(size_t)b * G_ + tid];
  float* ab = Abuf + (size_t)b * (DM_ * G_);
  #pragma unroll
  for (int k = 0; k < DM_; ++k) ab[k * G_ + tid] = h[k] * gj;
}

// g_gen = clip(LN(g + tmv), -1, 1)
__global__ __launch_bounds__(128) void ggen_ln_k(
    const float* __restrict__ g, const float* __restrict__ tmv, float* __restrict__ out)
{
  __shared__ float red[128];
  int b = blockIdx.x, tid = threadIdx.x;
  float xv = g[(size_t)b * G_ + tid] + tmv[(size_t)b * G_ + tid];
  float m = blk_sum(xv, red) * (1.f / G_);
  float c = xv - m;
  float v = blk_sum(c * c, red) * (1.f / G_);
  float y = c * rsqrtf(v + 1e-6f);
  out[(size_t)b * G_ + tid] = fminf(1.f, fmaxf(-1.f, y));
}

// g_new = clip(LN(gg + delta*(xm - gg)), -1, 1)
__global__ __launch_bounds__(128) void gnew_ln_k(
    const float* __restrict__ gg, const float* __restrict__ delta,
    const float* __restrict__ xm, float* __restrict__ out)
{
  __shared__ float red[128];
  int b = blockIdx.x, tid = threadIdx.x;
  size_t idx = (size_t)b * G_ + tid;
  float gv0 = gg[idx];
  float xv = gv0 + delta[idx] * (xm[idx] - gv0);
  float m = blk_sum(xv, red) * (1.f / G_);
  float c = xv - m;
  float v = blk_sum(c * c, red) * (1.f / G_);
  float y = c * rsqrtf(v + 1e-6f);
  out[idx] = fminf(1.f, fmaxf(-1.f, y));
}

// din = concat([x_mem_g, g_gen, ent, err, zero-pad])
__global__ __launch_bounds__(128) void dinb_k(
    const float* __restrict__ xm, const float* __restrict__ gg,
    const float* __restrict__ ent, const float* __restrict__ err, float* __restrict__ din)
{
  int b = blockIdx.x, tid = threadIdx.x;
  float* dr = din + (size_t)b * DINP_;
  for (int c = tid; c < DINP_; c += 128) {
    float v;
    if (c < G_)               v = xm[(size_t)b * G_ + c];
    else if (c < 2 * G_)      v = gg[(size_t)b * G_ + (c - G_)];
    else if (c == 2 * G_)     v = ent[b];
    else if (c == 2 * G_ + 1) v = err[b];
    else                      v = 0.f;
    dr[c] = v;
  }
}

// xk[b,i,:] = vis*x_t ; gv[b,i,:] = vis*g_mem
__global__ __launch_bounds__(128) void memupd_k(
    const float* __restrict__ x_t, const float* __restrict__ g_mem,
    const float* __restrict__ visrow, float* __restrict__ xk, float* __restrict__ gv, int i)
{
  int b = blockIdx.x, tid = threadIdx.x;
  float vv = visrow[b];
  float* xkr = xk + ((size_t)b * S_ + i) * SS_;
  const float* xr = x_t + (size_t)b * SS_;
  for (int c = tid; c < SS_; c += 128) xkr[c] = vv * xr[c];
  float* gvr = gv + ((size_t)b * S_ + i) * PHASE_;
  const float* gr = g_mem + (size_t)b * PHASE_;
  for (int c = tid; c < PHASE_; c += 128) gvr[c] = vv * gr[c];
}

__global__ void copyf_k(const float* __restrict__ s, float* __restrict__ d, int n) {
  for (int i = blockIdx.x * blockDim.x + threadIdx.x; i < n; i += gridDim.x * blockDim.x)
    d[i] = s[i];
}

// transpose+convert: src (K x N, f32) -> dst (N x Kpad, bf16), zero-padded rows
__global__ void wtrans_k(const float* __restrict__ src, unsigned short* __restrict__ dst,
                         int K, int N, int Kpad)
{
  int idx = blockIdx.x * blockDim.x + threadIdx.x;
  if (idx >= N * Kpad) return;
  int n = idx / Kpad, k = idx - n * Kpad;
  dst[idx] = (k < K) ? f2bf(src[(size_t)k * N + n]) : (unsigned short)0;
}

// special layout for the t_mat weight: dst[i*4096 + k*128 + j] = w2[k, i*128 + j]
__global__ void w2trans_k(const float* __restrict__ src, unsigned short* __restrict__ dst)
{
  int idx = blockIdx.x * blockDim.x + threadIdx.x;
  if (idx >= G_ * DM_ * G_) return;
  int n = idx >> 12;          // output row i (0..127)
  int kk = idx & 4095;        // k*128 + j
  int k = kk >> 7, j = kk & 127;
  dst[idx] = f2bf(src[(size_t)k * (G_ * G_) + n * G_ + j]);
}

// ---------------- host side ----------------
static void launch_gemm(hipStream_t stream, const float* A, const unsigned short* WT,
                        const float* bias, float* C, int N, int K, int act)
{
  dim3 grid(N / 64, B_ / 128), blk(256);
  if (act == 0)      gemm_bf16<0><<<grid, blk, 0, stream>>>(A, WT, bias, C, N, K);
  else if (act == 1) gemm_bf16<1><<<grid, blk, 0, stream>>>(A, WT, bias, C, N, K);
  else               gemm_bf16<2><<<grid, blk, 0, stream>>>(A, WT, bias, C, N, K);
}

extern "C" void kernel_launch(void* const* d_in, const int* in_sizes, int n_in,
                              void* d_out, int out_size, void* d_ws, size_t ws_size,
                              hipStream_t stream)
{
  (void)in_sizes; (void)n_in; (void)out_size; (void)ws_size;

  const float* x    = (const float*)d_in[0];
  const float* dct  = (const float*)d_in[1];
  const float* g0   = (const float*)d_in[2];
  const float* vis  = (const float*)d_in[3];
  const float* w1   = (const float*)d_in[4];
  const float* w2   = (const float*)d_in[5];
  const float* pg   = (const float*)d_in[6];
  const float* muw1 = (const float*)d_in[7];
  const float* mub1 = (const float*)d_in[8];
  const float* muw2 = (const float*)d_in[9];
  const float* mub2 = (const float*)d_in[10];
  const float* dw1  = (const float*)d_in[11];
  const float* db1  = (const float*)d_in[12];
  const float* dw2  = (const float*)d_in[13];
  const float* db2  = (const float*)d_in[14];
  const float* xw1  = (const float*)d_in[15];
  const float* xb1  = (const float*)d_in[16];
  const float* xw2  = (const float*)d_in[17];
  const float* xb2  = (const float*)d_in[18];

  float* outxg  = (float*)d_out;
  float* outxgt = outxg + (size_t)S_ * B_ * SS_;

  char* base = (char*)d_ws;
  size_t off = 0;
  auto alloc = [&](size_t bytes) -> void* {
    void* p = base + off;
    off += (bytes + 255) & ~(size_t)255;
    return p;
  };

  // f32 state + activations
  float* xk     = (float*)alloc((size_t)B_ * S_ * SS_ * 4);
  float* gvm    = (float*)alloc((size_t)B_ * S_ * PHASE_ * 4);
  float* g_cur  = (float*)alloc((size_t)B_ * G_ * 4);
  float* g_gen  = (float*)alloc((size_t)B_ * G_ * 4);
  float* tmv    = (float*)alloc((size_t)B_ * G_ * 4);
  float* Abuf   = (float*)alloc((size_t)B_ * DM_ * G_ * 4);
  float* r_x2g  = (float*)alloc((size_t)B_ * PHASE_ * 4);
  float* x_retr = (float*)alloc((size_t)B_ * SS_ * 4);
  float* entb   = (float*)alloc((size_t)B_ * 4);
  float* errb   = (float*)alloc((size_t)B_ * 4);
  float* t_mu   = (float*)alloc((size_t)B_ * G2_ * 4);
  float* x_memg = (float*)alloc((size_t)B_ * G_ * 4);
  float* dinbuf = (float*)alloc((size_t)B_ * DINP_ * 4);
  float* hid_d  = (float*)alloc((size_t)B_ * DH_ * 4);
  float* delta  = (float*)alloc((size_t)B_ * G_ * 4);
  float* g_mem  = (float*)alloc((size_t)B_ * PHASE_ * 4);
  float* gt_mem = (float*)alloc((size_t)B_ * PHASE_ * 4);
  float* r_g2x  = (float*)alloc((size_t)B_ * SS_ * 4);
  float* r_gt2x = (float*)alloc((size_t)B_ * SS_ * 4);
  float* mh     = (float*)alloc((size_t)B_ * SH_ * 4);
  float* mh2    = (float*)alloc((size_t)B_ * SH_ * 4);

  // bf16 weights (N-major [n][k], K padded to /32)
  unsigned short* w2t  = (unsigned short*)alloc((size_t)G_ * DM_ * G_ * 2);   // 128 x 4096
  unsigned short* mu1t = (unsigned short*)alloc((size_t)G2_ * PHASE_ * 2);    // 256 x 256
  unsigned short* mu2t = (unsigned short*)alloc((size_t)G_ * G2_ * 2);        // 128 x 256
  unsigned short* d1t  = (unsigned short*)alloc((size_t)DH_ * DINP_ * 2);     // 1280 x 288
  unsigned short* d2t  = (unsigned short*)alloc((size_t)G_ * DH_ * 2);        // 128 x 1280
  unsigned short* pgt  = (unsigned short*)alloc((size_t)PHASE_ * G_ * 2);     // 256 x 128
  unsigned short* xp1t = (unsigned short*)alloc((size_t)SH_ * SS_ * 2);       // 1024 x 512
  unsigned short* xp2t = (unsigned short*)alloc((size_t)SS_ * SH_ * 2);       // 512 x 1024

  const int TP = 256;
  auto cdiv = [](long long a, long long b) { return (unsigned)((a + b - 1) / b); };

  // ---- prologue: weight conversion + g init (per launch, deterministic) ----
  w2trans_k<<<cdiv((long long)G_ * DM_ * G_, TP), TP, 0, stream>>>(w2, w2t);
  wtrans_k<<<cdiv((long long)G2_ * PHASE_, TP), TP, 0, stream>>>(muw1, mu1t, PHASE_, G2_, PHASE_);
  wtrans_k<<<cdiv((long long)G_ * G2_, TP), TP, 0, stream>>>(muw2, mu2t, G2_, G_, G2_);
  wtrans_k<<<cdiv((long long)DH_ * DINP_, TP), TP, 0, stream>>>(dw1, d1t, 2 * G_ + 2, DH_, DINP_);
  wtrans_k<<<cdiv((long long)G_ * DH_, TP), TP, 0, stream>>>(dw2, d2t, DH_, G_, DH_);
  wtrans_k<<<cdiv((long long)PHASE_ * G_, TP), TP, 0, stream>>>(pg, pgt, G_, PHASE_, G_);
  wtrans_k<<<cdiv((long long)SH_ * SS_, TP), TP, 0, stream>>>(xw1, xp1t, SS_, SH_, SS_);
  wtrans_k<<<cdiv((long long)SS_ * SH_, TP), TP, 0, stream>>>(xw2, xp2t, SH_, SS_, SH_);
  copyf_k<<<128, 256, 0, stream>>>(g0, g_cur, B_ * G_);
  // xk/gv never need zeroing — slot s is only read when s < i, after being written.

  // ---- sequential scan ----
  for (int i = 0; i < S_; ++i) {
    const float* x_t = x   + (size_t)i * B_ * SS_;
    const float* d_t = dct + (size_t)i * B_ * NA_;
    const float* v_t = vis + (size_t)i * B_;

    // g_gen = clip(LN(g + t_mat@g)) via factored GEMM (B x 4096) * (4096 x 128)
    phaseA_k<<<B_, 128, 0, stream>>>(d_t, w1, g_cur, Abuf);
    launch_gemm(stream, Abuf, w2t, nullptr, tmv, G_, DM_ * G_, 0);
    ggen_ln_k<<<B_, 128, 0, stream>>>(g_cur, tmv, g_gen);

    // retrievals (per-batch matvecs, L2-resident state)
    retrieve_k<SS_, PHASE_, true, false><<<B_, 128, 0, stream>>>(
        x_t, xk, gvm, r_x2g, entb, nullptr, nullptr, i);
    retrieve_k<PHASE_, SS_, false, true><<<B_, 128, 0, stream>>>(
        r_x2g, gvm, xk, x_retr, nullptr, errb, x_t, i);

    // p2g_mu MLP
    launch_gemm(stream, r_x2g, mu1t, mub1, t_mu, G2_, PHASE_, 1);
    launch_gemm(stream, t_mu, mu2t, mub2, x_memg, G_, G2_, 0);

    // delta MLP (din padded 258 -> 288)
    dinb_k<<<B_, 128, 0, stream>>>(x_memg, g_gen, entb, errb, dinbuf);
    launch_gemm(stream, dinbuf, d1t, db1, hid_d, DH_, DINP_, 1);
    launch_gemm(stream, hid_d, d2t, db2, delta, G_, DH_, 2);

    // g_new (stored back into g_cur for next step)
    gnew_ln_k<<<B_, 128, 0, stream>>>(g_gen, delta, x_memg, g_cur);

    // projections
    launch_gemm(stream, g_cur, pgt, nullptr, g_mem, PHASE_, G_, 0);
    launch_gemm(stream, g_gen, pgt, nullptr, gt_mem, PHASE_, G_, 0);

    // g->x retrievals
    retrieve_k<PHASE_, SS_, false, false><<<B_, 128, 0, stream>>>(
        g_mem, gvm, xk, r_g2x, nullptr, nullptr, nullptr, i);
    retrieve_k<PHASE_, SS_, false, false><<<B_, 128, 0, stream>>>(
        gt_mem, gvm, xk, r_gt2x, nullptr, nullptr, nullptr, i);

    // mlp_x, outputs written straight to d_out (layout (2,S,B,SS))
    launch_gemm(stream, r_g2x, xp1t, xb1, mh, SH_, SS_, 1);
    launch_gemm(stream, mh, xp2t, xb2, outxg + (size_t)i * B_ * SS_, SS_, SH_, 0);
    launch_gemm(stream, r_gt2x, xp1t, xb1, mh2, SH_, SS_, 1);
    launch_gemm(stream, mh2, xp2t, xb2, outxgt + (size_t)i * B_ * SS_, SS_, SH_, 0);

    // memory write for slot i
    memupd_k<<<B_, 128, 0, stream>>>(x_t, g_mem, v_t, xk, gvm, i);
  }
}
